// NodePreTrans_18073222381669
// MI455X (gfx1250) — matrix operand order repaired and checked by reading, not compiled, not run on hardware
//
#include <hip/hip_runtime.h>

// ---------------------------------------------------------------------------
// e3nn-style fused tensor product for MI455X (gfx1250).
// bf16 WMMA (v_wmma_f32_16x16x32_bf16) with f32 accumulation.
// v3: 32 nodes/WG, 8 wave32. B fragments double-buffered (hide L2 latency),
// and all j-replicated GEMMs (p2, p4/p5, out_1o, out_1e) fused so one B
// fragment feeds 6 WMMAs (3 vector components x 2 row tiles). Weight (L2)
// traffic drops to ~one weight-set pass per WG (~0.65MB), balancing the
// 333MB HBM floor instead of dominating it.
// ---------------------------------------------------------------------------

typedef __attribute__((ext_vector_type(16))) __bf16 bf16x16;
typedef __attribute__((ext_vector_type(8)))  float  floatx8;

union Frag {
  bf16x16 v;
  unsigned u[8];
};

__device__ __forceinline__ unsigned short f2bf(float f) {
  union { float f; unsigned u; } x; x.f = f;
  unsigned r = x.u + 0x7FFFu + ((x.u >> 16) & 1u);   // round-to-nearest-even
  return (unsigned short)(r >> 16);
}
__device__ __forceinline__ float bf2f(unsigned short b) {
  union { float f; unsigned u; } x; x.u = ((unsigned)b) << 16;
  return x.f;
}
__device__ __forceinline__ floatx8 fzero() {
  floatx8 z = {0.f, 0.f, 0.f, 0.f, 0.f, 0.f, 0.f, 0.f};
  return z;
}
__device__ __forceinline__ void pack_b(Frag& fb, uint4 b0, uint4 b1) {
  fb.u[0] = b0.x; fb.u[1] = b0.y; fb.u[2] = b0.z; fb.u[3] = b0.w;
  fb.u[4] = b1.x; fb.u[5] = b1.y; fb.u[6] = b1.z; fb.u[7] = b1.w;
}
// A lane layout (contiguous k order, matching B):
//   lanes 0-15 : M=lane,    K in {k0..k0+15}
//   lanes 16-31: M=lane-16, K in {k0+16..k0+31}
__device__ __forceinline__ void load_a(Frag& fa, const unsigned short* ar) {
#pragma unroll
  for (int p = 0; p < 4; ++p) {
    fa.u[p]     = *(const unsigned*)(ar + 2 * p);
    fa.u[4 + p] = *(const unsigned*)(ar + 8 + 2 * p);
  }
}

// Two 16x16 f32 tiles of D = A(32xK) x B(Kx16); B double-buffered from global.
__device__ __forceinline__ void gemm_tile2(
    const unsigned short* __restrict__ A, int lda,
    const unsigned short* __restrict__ B, int ldb,
    int nTile, int K, int lane, floatx8& acc0, floatx8& acc1) {
  const int m  = lane & 15;
  const int hl = lane >> 4;
  const unsigned short* a0   = A + m * lda + hl * 16;
  const unsigned short* brow = B + (nTile * 16 + m) * ldb + hl * 16;
  uint4 nb0 = *(const uint4*)(brow);
  uint4 nb1 = *(const uint4*)(brow + 8);
  for (int k0 = 0; k0 < K; k0 += 32) {
    uint4 b0 = nb0, b1 = nb1;
    if (k0 + 32 < K) {
      nb0 = *(const uint4*)(brow + k0 + 32);
      nb1 = *(const uint4*)(brow + k0 + 40);
    }
    Frag fb; pack_b(fb, b0, b1);
#pragma unroll
    for (int sub = 0; sub < 2; ++sub) {
      Frag fa; load_a(fa, a0 + sub * 16 * lda + k0);
      floatx8& acc = sub ? acc1 : acc0;
      acc = __builtin_amdgcn_wmma_f32_16x16x32_bf16(false, fa.v, false, fb.v,
                                                    (short)0, acc, false, false);
    }
  }
}

// Six 16x16 f32 tiles: acc[j][sub] += A_j(32xK) x B(Kx16) for j=0..2,
// sharing each (double-buffered) B fragment across all 6 WMMAs.
__device__ __forceinline__ void gemm_j3_tile2(
    const unsigned short* __restrict__ A, int jstride, int lda,
    const unsigned short* __restrict__ B, int ldb,
    int nTile, int K, int lane, floatx8 (&acc)[3][2]) {
  const int m  = lane & 15;
  const int hl = lane >> 4;
  const unsigned short* a0   = A + m * lda + hl * 16;
  const unsigned short* brow = B + (nTile * 16 + m) * ldb + hl * 16;
  uint4 nb0 = *(const uint4*)(brow);
  uint4 nb1 = *(const uint4*)(brow + 8);
  for (int k0 = 0; k0 < K; k0 += 32) {
    uint4 b0 = nb0, b1 = nb1;
    if (k0 + 32 < K) {
      nb0 = *(const uint4*)(brow + k0 + 32);
      nb1 = *(const uint4*)(brow + k0 + 40);
    }
    Frag fb; pack_b(fb, b0, b1);
#pragma unroll
    for (int j = 0; j < 3; ++j)
#pragma unroll
      for (int sub = 0; sub < 2; ++sub) {
        Frag fa; load_a(fa, a0 + j * jstride + sub * 16 * lda + k0);
        acc[j][sub] = __builtin_amdgcn_wmma_f32_16x16x32_bf16(
            false, fa.v, false, fb.v, (short)0, acc[j][sub], false, false);
      }
  }
}

// ---- weight preprocessing ----
__global__ void __launch_bounds__(256) cvt_copy(const float* __restrict__ src,
                                                unsigned short* __restrict__ dst,
                                                int n) {
  int i = blockIdx.x * 256 + threadIdx.x;
  if (i < n) dst[i] = f2bf(src[i]);
}
// src (rows, cols) f32 -> dst (cols, rows) bf16
__global__ void __launch_bounds__(256) cvt_transpose(const float* __restrict__ src,
                                                     unsigned short* __restrict__ dst,
                                                     int rows, int cols) {
  int i = blockIdx.x * 256 + threadIdx.x;
  if (i < rows * cols) {
    int r = i / cols, c = i % cols;
    dst[c * rows + r] = f2bf(src[i]);
  }
}

// ---- fused kernel: 32 nodes per workgroup, 8 waves, 160KB dynamic LDS ----
__global__ void __launch_bounds__(256)
fused_tp_kernel(const float* __restrict__ node_feat,
                float* __restrict__ out,
                const unsigned short* __restrict__ wbase,
                int Ntot) {
  // bf16 weight regions inside d_ws (ushort offsets)
  const unsigned short* w00  = wbase;            // (256,256)  B[k,n]=w[n,k]
  const unsigned short* w011 = wbase + 65536;    // (256,128)
  const unsigned short* w101 = wbase + 98304;    // (128,256)
  const unsigned short* w110 = wbase + 131072;   // (128,128)
  const unsigned short* w111 = wbase + 147456;   // (128,128)
  const unsigned short* W0eT = wbase + 163840;   // (256,384) pre-transposed
  const unsigned short* W1oT = wbase + 262144;   // (128,384) pre-transposed
  const unsigned short* W1eT = wbase + 311296;   // (128,128) pre-transposed

  extern __shared__ unsigned short smem[];
  // LDS map (ushort offsets), 32-row tiles:
  //   sA      : 0             32x256
  //   vA[j]   : 8192+j*4096   32x128
  //   tp0e    : 20480         32x384  ([p1|p4])
  //   tp1o[j] : 32768+j*12288 32x384  ([p2|p3])
  //   p5A[j]  : 69632+j*4096  32x128
  unsigned short* sA   = smem;
  unsigned short* tp0e = smem + 20480;

  const int tid  = threadIdx.x;
  const int lane = tid & 31;
  const int wv   = tid >> 5;
  const int m    = lane & 15;
  const int hl   = lane >> 4;
  const int z0   = blockIdx.x * 32;
  const int rows = (Ntot - z0 < 32) ? (Ntot - z0) : 32;

  const float C000 = 0.0625f;               // 1/sqrt(256)
  const float C011 = 0.08838834764831845f;  // 1/sqrt(128)
  const float C101 = 0.0625f;               // 1/sqrt(256)
  const float C110 = 0.05103103630798288f;  // 1/sqrt(384)
  const float C111 = 0.0625f;               // 1/16
  const float S384 = 0.05103103630798288f;  // 1/sqrt(384)
  const float S128 = 0.08838834764831845f;  // 1/sqrt(128)

  // ---- stage node tile to LDS as bf16 (zero-fill tail rows) ----
  for (int idx = tid; idx < 32 * 640; idx += 256) {
    int row = idx / 640, c = idx % 640;
    float val = (row < rows) ? node_feat[(size_t)(z0 + row) * 640 + c] : 0.f;
    if (c < 256) {
      sA[row * 256 + c] = f2bf(val);
    } else {
      int cc = c - 256;
      int u = cc / 3, j = cc % 3;
      (smem + 8192 + j * 4096)[row * 128 + u] = f2bf(val);
    }
  }
  __syncthreads();

  // ================= Phase A: layer-1 GEMMs + elementwise =================
  // p1 = C000 * s .* (s @ w00^T)  -> tp0e cols [0,256)
  for (int t = wv; t < 16; t += 8) {
    floatx8 acc0 = fzero(), acc1 = fzero();
    gemm_tile2(sA, 256, w00, 256, t, 256, lane, acc0, acc1);
    const int col = t * 16 + m;
#pragma unroll
    for (int sub = 0; sub < 2; ++sub) {
      const floatx8& acc = sub ? acc1 : acc0;
#pragma unroll
      for (int r = 0; r < 8; ++r) {
        int zr = r + 8 * hl + 16 * sub;
        float s = bf2f(sA[zr * 256 + col]);
        tp0e[zr * 384 + col] = f2bf(C000 * s * acc[r]);
      }
    }
  }

  // p2_j = C011 * s .* (V_j @ w011^T) -> tp1o[j] cols [0,256)
  // one w011 B-stream feeds all 3 j
  for (int t = wv; t < 16; t += 8) {
    floatx8 acc[3][2];
#pragma unroll
    for (int j = 0; j < 3; ++j) { acc[j][0] = fzero(); acc[j][1] = fzero(); }
    gemm_j3_tile2(smem + 8192, 4096, 128, w011, 128, t, 128, lane, acc);
    const int col = t * 16 + m;
#pragma unroll
    for (int sub = 0; sub < 2; ++sub)
#pragma unroll
      for (int r = 0; r < 8; ++r) {
        int zr = r + 8 * hl + 16 * sub;
        float s = C011 * bf2f(sA[zr * 256 + col]);
#pragma unroll
        for (int j = 0; j < 3; ++j)
          (smem + 32768 + j * 12288)[zr * 384 + col] = f2bf(s * acc[j][sub][r]);
      }
  }

  // T3 = s @ w101^T ; p3_j = C101 * V_j .* T3 -> tp1o[j] cols [256,384)
  for (int t = wv; t < 8; t += 8) {
    floatx8 acc0 = fzero(), acc1 = fzero();
    gemm_tile2(sA, 256, w101, 256, t, 256, lane, acc0, acc1);
    const int col = t * 16 + m;
#pragma unroll
    for (int sub = 0; sub < 2; ++sub) {
      const floatx8& acc = sub ? acc1 : acc0;
#pragma unroll
      for (int r = 0; r < 8; ++r) {
        int zr = r + 8 * hl + 16 * sub;
        float T3 = C101 * acc[r];
#pragma unroll
        for (int j = 0; j < 3; ++j) {
          float vj = bf2f((smem + 8192 + j * 4096)[zr * 128 + col]);
          (smem + 32768 + j * 12288)[zr * 384 + 256 + col] = f2bf(vj * T3);
        }
      }
    }
  }

  // p4 = C110 * sum_j V_j .* (V_j @ w110^T)      -> tp0e cols [256,384)
  // t5_j = V_j @ w111^T ; p5 = C111 * (v x t5)   -> p5A[j]
  // each of w110 / w111 streamed once, feeding all 3 j
  for (int t = wv; t < 8; t += 8) {
    const int col = t * 16 + m;
    float p4acc[2][8];
#pragma unroll
    for (int sub = 0; sub < 2; ++sub)
#pragma unroll
      for (int r = 0; r < 8; ++r) p4acc[sub][r] = 0.f;
    {
      floatx8 a4[3][2];
#pragma unroll
      for (int j = 0; j < 3; ++j) { a4[j][0] = fzero(); a4[j][1] = fzero(); }
      gemm_j3_tile2(smem + 8192, 4096, 128, w110, 128, t, 128, lane, a4);
#pragma unroll
      for (int j = 0; j < 3; ++j)
#pragma unroll
        for (int sub = 0; sub < 2; ++sub)
#pragma unroll
          for (int r = 0; r < 8; ++r) {
            int zr = r + 8 * hl + 16 * sub;
            p4acc[sub][r] +=
                bf2f((smem + 8192 + j * 4096)[zr * 128 + col]) * a4[j][sub][r];
          }
    }
    floatx8 t5[3][2];
#pragma unroll
    for (int j = 0; j < 3; ++j) { t5[j][0] = fzero(); t5[j][1] = fzero(); }
    gemm_j3_tile2(smem + 8192, 4096, 128, w111, 128, t, 128, lane, t5);
#pragma unroll
    for (int sub = 0; sub < 2; ++sub)
#pragma unroll
      for (int r = 0; r < 8; ++r) {
        int zr = r + 8 * hl + 16 * sub;
        tp0e[zr * 384 + 256 + col] = f2bf(C110 * p4acc[sub][r]);
        float vx = bf2f((smem + 8192 + 0 * 4096)[zr * 128 + col]);
        float vy = bf2f((smem + 8192 + 1 * 4096)[zr * 128 + col]);
        float vz = bf2f((smem + 8192 + 2 * 4096)[zr * 128 + col]);
        float tx = t5[0][sub][r], ty = t5[1][sub][r], tz = t5[2][sub][r];
        (smem + 69632 + 0 * 4096)[zr * 128 + col] = f2bf(C111 * (vy * tz - vz * ty));
        (smem + 69632 + 1 * 4096)[zr * 128 + col] = f2bf(C111 * (vz * tx - vx * tz));
        (smem + 69632 + 2 * 4096)[zr * 128 + col] = f2bf(C111 * (vx * ty - vy * tx));
      }
  }
  __syncthreads();

  // ================= Phase B: layer-2 GEMMs -> global output ==============
  // out_0e = [p1|p4] @ W0e / sqrt(384)     -> out cols [0,256)
  for (int t = wv; t < 16; t += 8) {
    floatx8 acc0 = fzero(), acc1 = fzero();
    gemm_tile2(tp0e, 384, W0eT, 384, t, 384, lane, acc0, acc1);
    const int o = t * 16 + m;
#pragma unroll
    for (int sub = 0; sub < 2; ++sub) {
      const floatx8& acc = sub ? acc1 : acc0;
#pragma unroll
      for (int r = 0; r < 8; ++r) {
        int zr = r + 8 * hl + 16 * sub;
        if (z0 + zr < Ntot)
          out[(size_t)(z0 + zr) * 1024 + o] = acc[r] * S384;
      }
    }
  }

  // out_1o[z,o,j] = ([p2_j|p3_j] @ W1o)/sqrt(384) -> out cols 256 + 3*o + j
  // one W1oT B-stream feeds all 3 j
  for (int t = wv; t < 8; t += 8) {
    floatx8 acc[3][2];
#pragma unroll
    for (int j = 0; j < 3; ++j) { acc[j][0] = fzero(); acc[j][1] = fzero(); }
    gemm_j3_tile2(smem + 32768, 12288, 384, W1oT, 384, t, 384, lane, acc);
    const int o = t * 16 + m;
#pragma unroll
    for (int sub = 0; sub < 2; ++sub)
#pragma unroll
      for (int r = 0; r < 8; ++r) {
        int zr = r + 8 * hl + 16 * sub;
        if (z0 + zr < Ntot) {
#pragma unroll
          for (int j = 0; j < 3; ++j)
            out[(size_t)(z0 + zr) * 1024 + 256 + o * 3 + j] = acc[j][sub][r] * S384;
        }
      }
  }

  // out_1e[z,o,j] = (p5_j @ W1e)/sqrt(128) -> out cols 640 + 3*o + j
  for (int t = wv; t < 8; t += 8) {
    floatx8 acc[3][2];
#pragma unroll
    for (int j = 0; j < 3; ++j) { acc[j][0] = fzero(); acc[j][1] = fzero(); }
    gemm_j3_tile2(smem + 69632, 4096, 128, W1eT, 128, t, 128, lane, acc);
    const int o = t * 16 + m;
#pragma unroll
    for (int sub = 0; sub < 2; ++sub)
#pragma unroll
      for (int r = 0; r < 8; ++r) {
        int zr = r + 8 * hl + 16 * sub;
        if (z0 + zr < Ntot) {
#pragma unroll
          for (int j = 0; j < 3; ++j)
            out[(size_t)(z0 + zr) * 1024 + 640 + o * 3 + j] = acc[j][sub][r] * S128;
        }
      }
  }
}

extern "C" void kernel_launch(void* const* d_in, const int* in_sizes, int n_in,
                              void* d_out, int out_size, void* d_ws, size_t ws_size,
                              hipStream_t stream) {
  const float* node_feat = (const float*)d_in[0];
  const float* w_00_0 = (const float*)d_in[1];
  const float* w_01_1 = (const float*)d_in[2];
  const float* w_10_1 = (const float*)d_in[3];
  const float* w_11_0 = (const float*)d_in[4];
  const float* w_11_1 = (const float*)d_in[5];
  const float* W_0e   = (const float*)d_in[6];
  const float* W_1o   = (const float*)d_in[7];
  const float* W_1e   = (const float*)d_in[8];
  float* out = (float*)d_out;
  unsigned short* wb = (unsigned short*)d_ws;
  const int N = in_sizes[0] / 640;   // 50000

  auto nb = [](int n) { return (n + 255) / 256; };
  // direct bf16 conversions (layer-1 weights: used as B[k,n]=w[n,k], k-contig)
  cvt_copy<<<nb(65536), 256, 0, stream>>>(w_00_0, wb + 0,      65536);
  cvt_copy<<<nb(32768), 256, 0, stream>>>(w_01_1, wb + 65536,  32768);
  cvt_copy<<<nb(32768), 256, 0, stream>>>(w_10_1, wb + 98304,  32768);
  cvt_copy<<<nb(16384), 256, 0, stream>>>(w_11_0, wb + 131072, 16384);
  cvt_copy<<<nb(16384), 256, 0, stream>>>(w_11_1, wb + 147456, 16384);
  // layer-2 weights transposed so B[k,n] = WT[n,k] is k-contiguous
  cvt_transpose<<<nb(98304), 256, 0, stream>>>(W_0e, wb + 163840, 384, 256);
  cvt_transpose<<<nb(49152), 256, 0, stream>>>(W_1o, wb + 262144, 384, 128);
  cvt_transpose<<<nb(16384), 256, 0, stream>>>(W_1e, wb + 311296, 128, 128);

  // 32 nodes per workgroup, 8 wave32, 160KB dynamic LDS
  const int nblk = (N + 31) / 32;
  fused_tp_kernel<<<nblk, 256, 163840, stream>>>(node_feat, out, wb, N);
}
